// ConformerBlock_82257213653608
// MI455X (gfx1250) — matrix operand order repaired and checked by reading, not compiled, not run on hardware
//
#include <hip/hip_runtime.h>
#include <math.h>

// ---------------------------------------------------------------------------
// Conformer block for MI455X (gfx1250, wave32, WMMA 16x16x32 bf16,
// async global->LDS double-buffered GEMM pipeline, K=64 stages)
// ---------------------------------------------------------------------------
#define Dm   512
#define Hh   8
#define HD   64
#define CTXs 128
#define FFNm 2048
#define KSm  31
#define Bb   4
#define Ss   4096
#define Nt   (Bb * Ss)          // 16384 tokens
#define EPSm 1e-5f
#define LSTR 72                 // padded LDS row stride (ushorts): 144B = 36
                                // banks -> conflict-free fragment reads,
                                // still 16B-aligned for b128 access

typedef __attribute__((ext_vector_type(16))) __bf16          v16bf;
typedef __attribute__((ext_vector_type(8)))  float           v8f;
typedef __attribute__((ext_vector_type(8)))  unsigned short  us8;

union FragBF {                      // one WMMA 16-bit A/B fragment (8 VGPRs)
    v16bf v;
    us8   h[2];
};

__device__ __forceinline__ unsigned short f2bf(float f) {
    unsigned int u = __float_as_uint(f);
    unsigned int r = u + 0x7FFFu + ((u >> 16) & 1u);   // round-to-nearest-even
    return (unsigned short)(r >> 16);
}
__device__ __forceinline__ float bf2f(unsigned short s) {
    return __uint_as_float(((unsigned int)s) << 16);
}
__device__ __forceinline__ us8 ld_us8(const unsigned short* p) {
    return *(const us8*)p;
}
__device__ __forceinline__ v8f wmma_bf16(v16bf a, v16bf b, v8f c) {
    // D = A(16x32 bf16) x B(32x16 bf16) + C(16x16 f32)
    return __builtin_amdgcn_wmma_f32_16x16x32_bf16(
        false, a, false, b, (short)0, c, false, false);
}

// CDNA5 async copy: 16B global -> LDS, tracked by ASYNCcnt (no VGPR data).
__device__ __forceinline__ void async_ld_b128(unsigned lds_off,
                                              const unsigned short* g) {
    asm volatile("global_load_async_to_lds_b128 %0, %1, off"
                 :: "v"(lds_off), "v"((unsigned long long)(size_t)g)
                 : "memory");
}
__device__ __forceinline__ void wait_asynccnt0() {
    asm volatile("s_wait_asynccnt 0x0" ::: "memory");
}

// ---------------------------------------------------------------------------
// Weight transpose + fp32 -> bf16:  src[K,N] -> dst[N,K]
// ---------------------------------------------------------------------------
__global__ __launch_bounds__(256)
void transpose_to_bf16(const float* __restrict__ src,
                       unsigned short* __restrict__ dst, int K, int N) {
    size_t idx = (size_t)blockIdx.x * 256 + threadIdx.x;   // over dst [N,K]
    int n = (int)(idx / K);
    int k = (int)(idx % K);
    dst[idx] = f2bf(src[(size_t)k * N + n]);
}

// ---------------------------------------------------------------------------
// LayerNorm over D=512: one wave per token, 8 tokens per block
// ---------------------------------------------------------------------------
template <bool BF16OUT>
__global__ __launch_bounds__(256)
void layernorm_kernel(const float* __restrict__ x, const float* __restrict__ g,
                      const float* __restrict__ bv,
                      unsigned short* __restrict__ outB,
                      float* __restrict__ outF) {
    int tok  = blockIdx.x * 8 + (threadIdx.x >> 5);
    int lane = threadIdx.x & 31;
    const float* row = x + (size_t)tok * Dm;
    float v[16];
    float s = 0.f;
#pragma unroll
    for (int j = 0; j < 16; j++) { v[j] = row[lane + 32 * j]; s += v[j]; }
    for (int m = 1; m < 32; m <<= 1) s += __shfl_xor(s, m, 32);
    float mean = s * (1.f / Dm);
    float vs = 0.f;
#pragma unroll
    for (int j = 0; j < 16; j++) { float d = v[j] - mean; vs += d * d; }
    for (int m = 1; m < 32; m <<= 1) vs += __shfl_xor(vs, m, 32);
    float inv = rsqrtf(vs * (1.f / Dm) + EPSm);
#pragma unroll
    for (int j = 0; j < 16; j++) {
        int c = lane + 32 * j;
        float o = (v[j] - mean) * inv * g[c] + bv[c];
        if (BF16OUT) outB[(size_t)tok * Dm + c] = f2bf(o);
        else         outF[(size_t)tok * Dm + c] = o;
    }
}

// ---------------------------------------------------------------------------
// WMMA GEMM:  C[M,N] = A[M,K](bf16) * Bt[N,K](bf16)^T + bias, fused epilogue.
// Block = 8 waves; wave = 32(M) x 64(N); block tile = 128 x 128.
// A/B panels (128x64 bf16) staged to LDS via async copies, double-buffered:
// one barrier per 16 WMMAs/wave, next-panel DMA overlaps current math.
// ---------------------------------------------------------------------------
enum { EPI_GELU_BF16 = 0, EPI_RESID_F32 = 1, EPI_BF16 = 2,
       EPI_QK_HEAD = 3, EPI_V_HEADT = 4 };

__device__ __forceinline__ void stage_panels64(
    const unsigned short* __restrict__ A, const unsigned short* __restrict__ Bt,
    unsigned short* lsA, unsigned short* lsB,
    int mBase, int nBase, int kb, int K, int tid) {
#pragma unroll
    for (int rep = 0; rep < 4; rep++) {
        int id = tid + rep * 256;          // 0..1023 : 128 rows x 8 x 16B
        int r = id >> 3, c = id & 7;
        async_ld_b128((unsigned)(size_t)(lsA + r * LSTR + c * 8),
                      A + (size_t)(mBase + r) * K + kb + c * 8);
        async_ld_b128((unsigned)(size_t)(lsB + r * LSTR + c * 8),
                      Bt + (size_t)(nBase + r) * K + kb + c * 8);
    }
}

template <int EPI>
__global__ __launch_bounds__(256)
void gemm_bf16_kernel(const unsigned short* __restrict__ A,
                      const unsigned short* __restrict__ Bt,
                      const float* __restrict__ bias,
                      const float* __restrict__ resid, float scale,
                      float* __restrict__ outF,
                      unsigned short* __restrict__ outB,
                      int N, int K) {
    __shared__ __align__(16) unsigned short lsA[2][128 * LSTR];
    __shared__ __align__(16) unsigned short lsB[2][128 * LSTR];

    int tid  = threadIdx.x;
    int lane = tid & 31;
    int wid  = tid >> 5;
    int wm   = wid >> 1, wn = wid & 1;
    int half = lane >> 4, l16 = lane & 15;
    int mBase = blockIdx.y * 128;
    int nBase = blockIdx.x * 128;

    v8f acc[2][4];
#pragma unroll
    for (int im = 0; im < 2; im++)
#pragma unroll
        for (int in = 0; in < 4; in++) acc[im][in] = (v8f)0.f;

    stage_panels64(A, Bt, lsA[0], lsB[0], mBase, nBase, 0, K, tid);

    int buf = 0;
    for (int kb = 0; kb < K; kb += 64) {
        wait_asynccnt0();          // this wave's copies into lsX[buf] done
        __syncthreads();           // everyone's copies visible
        if (kb + 64 < K)           // overlap next panel DMA with WMMAs
            stage_panels64(A, Bt, lsA[buf ^ 1], lsB[buf ^ 1],
                           mBase, nBase, kb + 64, K, tid);

#pragma unroll
        for (int kk = 0; kk < 64; kk += 32) {
            FragBF a[2];
#pragma unroll
            for (int im = 0; im < 2; im++) {
                const unsigned short* ap =
                    lsA[buf] + (wm * 32 + 16 * im + l16) * LSTR + kk + half * 8;
                a[im].h[0] = *(const us8*)ap;        // K = kk+half*8 .. +7
                a[im].h[1] = *(const us8*)(ap + 16); // K = kk+16+half*8 .. +7
            }
            FragBF b[4];
#pragma unroll
            for (int in = 0; in < 4; in++) {
                const unsigned short* bp =
                    lsB[buf] + (wn * 64 + 16 * in + l16) * LSTR + kk + half * 8;
                b[in].h[0] = *(const us8*)bp;        // K = kk+half*8 .. +7
                b[in].h[1] = *(const us8*)(bp + 16); // K = kk+16+half*8 .. +7
            }
#pragma unroll
            for (int im = 0; im < 2; im++)
#pragma unroll
                for (int in = 0; in < 4; in++)
                    acc[im][in] = wmma_bf16(a[im].v, b[in].v, acc[im][in]);
        }
        buf ^= 1;
    }

    // Epilogue: C element (VGPR i) is (row m0+i+half*8, col n0+l16) per tile.
    int m0 = mBase + wm * 32;
    int n0 = nBase + wn * 64;
#pragma unroll
    for (int im = 0; im < 2; im++) {
#pragma unroll
        for (int in = 0; in < 4; in++) {
            int col = n0 + 16 * in + l16;
            float bvl = bias ? bias[col] : 0.f;
#pragma unroll
            for (int i = 0; i < 8; i++) {
                int row = m0 + 16 * im + i + half * 8;
                float v = acc[im][in][i] + bvl;
                if (EPI == EPI_GELU_BF16) {
                    v = 0.5f * v * (1.f + erff(v * 0.70710678118654752f));
                    outB[(size_t)row * N + col] = f2bf(v);
                } else if (EPI == EPI_RESID_F32) {
                    size_t idx = (size_t)row * N + col;
                    outF[idx] = resid[idx] + scale * v;
                } else if (EPI == EPI_BF16) {
                    outB[(size_t)row * N + col] = f2bf(v);
                } else if (EPI == EPI_QK_HEAD) {
                    int bz = row >> 12, s = row & (Ss - 1);
                    int h = col >> 6, d = col & 63;
                    outB[((((size_t)bz * Hh + h) * Ss) + s) * HD + d] = f2bf(v);
                } else { // EPI_V_HEADT : [B,H,d,s]
                    int bz = row >> 12, s = row & (Ss - 1);
                    int h = col >> 6, d = col & 63;
                    outB[((((size_t)bz * Hh + h) * HD) + d) * Ss + s] = f2bf(v);
                }
            }
        }
    }
}

// ---------------------------------------------------------------------------
// Windowed local attention, one workgroup per (block n, head h, batch b).
// Wave w handles 16 queries; flash-style online softmax over 3 key blocks.
// Q,K : [B,H,S,64] bf16 ; Vt : [B,H,64,S] bf16 ; AO : [token, 512] bf16
// ---------------------------------------------------------------------------
__global__ __launch_bounds__(256)
void local_attn_kernel(const unsigned short* __restrict__ Q,
                       const unsigned short* __restrict__ Kh,
                       const unsigned short* __restrict__ Vt,
                       const float* __restrict__ rel_bias,
                       unsigned short* __restrict__ AO) {
    __shared__ __align__(16) unsigned short ldsP[8 * 16 * 128];

    int nb = blockIdx.x, h = blockIdx.y, bz = blockIdx.z;
    int lane = threadIdx.x & 31, wid = threadIdx.x >> 5;
    int half = lane >> 4, l16 = lane & 15;

    const size_t ho = (size_t)bz * Hh + h;
    const unsigned short* Qh = Q  + ho * Ss * HD;
    const unsigned short* Kp = Kh + ho * Ss * HD;
    const unsigned short* Vp = Vt + ho * HD * Ss;

    int q0 = nb * CTXs + wid * 16;            // first query row of this wave

    FragBF qf[2];
#pragma unroll
    for (int kk = 0; kk < 2; kk++) {
        const unsigned short* qp = Qh + (size_t)(q0 + l16) * HD + kk * 32 + half * 8;
        qf[kk].h[0] = ld_us8(qp);
        qf[kk].h[1] = ld_us8(qp + 16);
    }

    v8f o[4];
#pragma unroll
    for (int t = 0; t < 4; t++) o[t] = (v8f)0.f;
    float Mr[8], Lr[8];
#pragma unroll
    for (int i = 0; i < 8; i++) { Mr[i] = -1e30f; Lr[i] = 0.f; }

    unsigned short* myP = ldsP + wid * (16 * 128);
    const float scl = 0.125f;                 // 64^-0.5

    for (int kbI = 0; kbI < 3; kbI++) {
        int key0 = nb * CTXs + (kbI - 1) * CTXs;   // global key base of block
        if (key0 < 0 || key0 >= Ss) continue;      // whole block off-sequence

        v8f sc[8];
#pragma unroll
        for (int t = 0; t < 8; t++) sc[t] = (v8f)0.f;
#pragma unroll
        for (int kk = 0; kk < 2; kk++) {
#pragma unroll
            for (int tg = 0; tg < 2; tg++) {       // groups of 4 key tiles
                FragBF kf[4];
#pragma unroll
                for (int j = 0; j < 4; j++) {
                    int t = tg * 4 + j;
                    const unsigned short* kp =
                        Kp + (size_t)(key0 + 16 * t + l16) * HD + kk * 32 + half * 8;
                    kf[j].h[0] = ld_us8(kp);
                    kf[j].h[1] = ld_us8(kp + 16);
                }
#pragma unroll
                for (int j = 0; j < 4; j++)
                    sc[tg * 4 + j] = wmma_bf16(qf[kk].v, kf[j].v, sc[tg * 4 + j]);
            }
        }
        // scale + relative-position bias + window mask (|rel| <= CTX)
#pragma unroll
        for (int t = 0; t < 8; t++) {
            int kj = kbI * CTXs + 16 * t + l16;    // window col 0..383
#pragma unroll
            for (int i = 0; i < 8; i++) {
                int qi = wid * 16 + i + half * 8;  // query pos in block 0..127
                int rel = kj - CTXs - qi;
                float v;
                if (rel >= -CTXs && rel <= CTXs)
                    v = sc[t][i] * scl + rel_bias[h * (2 * CTXs + 1) + rel + CTXs];
                else
                    v = -1e30f;
                sc[t][i] = v;
            }
        }
        // online softmax: per-row stats via intra-half shfl_xor reductions
#pragma unroll
        for (int i = 0; i < 8; i++) {
            float m = -1e30f;
#pragma unroll
            for (int t = 0; t < 8; t++) m = fmaxf(m, sc[t][i]);
            for (int msk = 1; msk < 16; msk <<= 1)
                m = fmaxf(m, __shfl_xor(m, msk, 32));
            float newM = fmaxf(Mr[i], m);
            float corr = __expf(Mr[i] - newM);
            Mr[i] = newM;
            float rs = 0.f;
#pragma unroll
            for (int t = 0; t < 8; t++) {
                float p = __expf(sc[t][i] - newM);
                sc[t][i] = p;
                rs += p;
            }
            for (int msk = 1; msk < 16; msk <<= 1)
                rs += __shfl_xor(rs, msk, 32);
            Lr[i] = Lr[i] * corr + rs;
#pragma unroll
            for (int tn = 0; tn < 4; tn++) o[tn][i] *= corr;
        }
        // stage P (16x128 bf16) through this wave's LDS slice (C -> A layout)
#pragma unroll
        for (int t = 0; t < 8; t++)
#pragma unroll
            for (int i = 0; i < 8; i++) {
                int row = i + half * 8;
                int col = 16 * t + l16;
                myP[row * 128 + col] = f2bf(sc[t][i]);
            }
        // P @ V (A = P from LDS, B = Vt rows, contiguous along keys)
#pragma unroll
        for (int kk = 0; kk < 4; kk++) {
            FragBF pf;
            const unsigned short* pp = myP + l16 * 128 + kk * 32 + half * 8;
            pf.h[0] = *(const us8*)pp;
            pf.h[1] = *(const us8*)(pp + 16);
            FragBF vf[4];
#pragma unroll
            for (int tn = 0; tn < 4; tn++) {
                int d = tn * 16 + l16;
                const unsigned short* vp =
                    Vp + (size_t)d * Ss + key0 + kk * 32 + half * 8;
                vf[tn].h[0] = ld_us8(vp);
                vf[tn].h[1] = ld_us8(vp + 16);
            }
#pragma unroll
            for (int tn = 0; tn < 4; tn++)
                o[tn] = wmma_bf16(pf.v, vf[tn].v, o[tn]);
        }
    }
    // normalize and scatter to [token, h*64+d]
#pragma unroll
    for (int tn = 0; tn < 4; tn++)
#pragma unroll
        for (int i = 0; i < 8; i++) {
            int row = i + half * 8;
            float v = o[tn][i] / Lr[i];
            int token = bz * Ss + q0 + row;
            int ch = h * HD + tn * 16 + l16;
            AO[(size_t)token * Dm + ch] = f2bf(v);
        }
}

// ---------------------------------------------------------------------------
// GLU: y = a * sigmoid(gate), g1 = [token, 2D]
// ---------------------------------------------------------------------------
__global__ __launch_bounds__(256)
void glu_kernel(const unsigned short* __restrict__ g1,
                unsigned short* __restrict__ y) {
    size_t idx = (size_t)blockIdx.x * 256 + threadIdx.x;
    size_t m = idx >> 9;
    int c = (int)(idx & (Dm - 1));
    float a = bf2f(g1[m * (2 * Dm) + c]);
    float g = bf2f(g1[m * (2 * Dm) + Dm + c]);
    y[idx] = f2bf(a / (1.f + __expf(-g)));
}

// ---------------------------------------------------------------------------
// Depthwise conv (KS=31, pad 15) + BatchNorm + SiLU ; y layout [token, c]
// ---------------------------------------------------------------------------
__global__ __launch_bounds__(256)
void dwconv_bn_silu_kernel(const unsigned short* __restrict__ y,
                           const float* __restrict__ w,
                           const float* __restrict__ wb,
                           const float* __restrict__ g,
                           const float* __restrict__ b,
                           const float* __restrict__ mn,
                           const float* __restrict__ vr,
                           unsigned short* __restrict__ out) {
    size_t idx = (size_t)blockIdx.x * 256 + threadIdx.x;
    int c = (int)(idx & (Dm - 1));
    size_t tok = idx >> 9;
    int s = (int)(tok & (Ss - 1));
    size_t bz = tok >> 12;
    float acc = 0.f;
#pragma unroll
    for (int t = 0; t < KSm; t++) {
        int s2 = s + t - KSm / 2;
        if (s2 >= 0 && s2 < Ss)
            acc += bf2f(y[((bz << 12) + s2) * Dm + c]) * w[c * KSm + t];
    }
    acc += wb[c];
    acc = (acc - mn[c]) * rsqrtf(vr[c] + EPSm) * g[c] + b[c];
    acc = acc / (1.f + __expf(-acc));     // SiLU
    out[idx] = f2bf(acc);
}

// ---------------------------------------------------------------------------
// Host-side orchestration
// ---------------------------------------------------------------------------
extern "C" void kernel_launch(void* const* d_in, const int* in_sizes, int n_in,
                              void* d_out, int out_size, void* d_ws,
                              size_t ws_size, hipStream_t stream) {
    (void)in_sizes; (void)n_in; (void)out_size; (void)ws_size;

    const float* x         = (const float*)d_in[0];
    const float* ff1_ln_g  = (const float*)d_in[1];
    const float* ff1_ln_b  = (const float*)d_in[2];
    const float* ff1_w1    = (const float*)d_in[3];
    const float* ff1_b1    = (const float*)d_in[4];
    const float* ff1_w2    = (const float*)d_in[5];
    const float* ff1_b2    = (const float*)d_in[6];
    const float* ff2_ln_g  = (const float*)d_in[7];
    const float* ff2_ln_b  = (const float*)d_in[8];
    const float* ff2_w1    = (const float*)d_in[9];
    const float* ff2_b1    = (const float*)d_in[10];
    const float* ff2_w2    = (const float*)d_in[11];
    const float* ff2_b2    = (const float*)d_in[12];
    const float* attn_ln_g = (const float*)d_in[13];
    const float* attn_ln_b = (const float*)d_in[14];
    const float* wq        = (const float*)d_in[15];
    const float* wk        = (const float*)d_in[16];
    const float* wv        = (const float*)d_in[17];
    const float* wo        = (const float*)d_in[18];
    const float* bq        = (const float*)d_in[19];
    const float* bk        = (const float*)d_in[20];
    const float* bvv       = (const float*)d_in[21];
    const float* bo        = (const float*)d_in[22];
    const float* rel_bias  = (const float*)d_in[23];
    const float* conv_ln_g = (const float*)d_in[24];
    const float* conv_ln_b = (const float*)d_in[25];
    const float* pw1_w     = (const float*)d_in[26];
    const float* pw1_b     = (const float*)d_in[27];
    const float* dw_w      = (const float*)d_in[28];
    const float* dw_b      = (const float*)d_in[29];
    const float* bn_g      = (const float*)d_in[30];
    const float* bn_b      = (const float*)d_in[31];
    const float* bn_m      = (const float*)d_in[32];
    const float* bn_v      = (const float*)d_in[33];
    const float* pw2_w     = (const float*)d_in[34];
    const float* pw2_b     = (const float*)d_in[35];
    const float* fin_ln_g  = (const float*)d_in[36];
    const float* fin_ln_b  = (const float*)d_in[37];

    char* ws = (char*)d_ws;
    size_t off = 0;
    auto alloc = [&](size_t bytes) -> char* {
        char* p = ws + off;
        off = (off + bytes + 255) & ~(size_t)255;
        return p;
    };

    unsigned short* wt_ff1_w1 = (unsigned short*)alloc((size_t)Dm * FFNm * 2);
    unsigned short* wt_ff1_w2 = (unsigned short*)alloc((size_t)FFNm * Dm * 2);
    unsigned short* wt_ff2_w1 = (unsigned short*)alloc((size_t)Dm * FFNm * 2);
    unsigned short* wt_ff2_w2 = (unsigned short*)alloc((size_t)FFNm * Dm * 2);
    unsigned short* wt_q      = (unsigned short*)alloc((size_t)Dm * Dm * 2);
    unsigned short* wt_k      = (unsigned short*)alloc((size_t)Dm * Dm * 2);
    unsigned short* wt_v      = (unsigned short*)alloc((size_t)Dm * Dm * 2);
    unsigned short* wt_o      = (unsigned short*)alloc((size_t)Dm * Dm * 2);
    unsigned short* wt_pw1    = (unsigned short*)alloc((size_t)Dm * 2 * Dm * 2);
    unsigned short* wt_pw2    = (unsigned short*)alloc((size_t)Dm * Dm * 2);

    unsigned short* xln  = (unsigned short*)alloc((size_t)Nt * Dm * 2);
    char*           Hbuf = alloc((size_t)Nt * FFNm * 2);        // 64 MB
    float*          X1   = (float*)alloc((size_t)Nt * Dm * 4);
    float*          X2   = (float*)alloc((size_t)Nt * Dm * 4);

    const size_t SLAB = (size_t)Nt * Dm * 2;                    // 16 MB
    unsigned short* Qb  = (unsigned short*)(Hbuf);
    unsigned short* Kb  = (unsigned short*)(Hbuf + SLAB);
    unsigned short* VTb = (unsigned short*)(Hbuf + 2 * SLAB);
    unsigned short* AOb = (unsigned short*)(Hbuf + 3 * SLAB);
    unsigned short* G1  = (unsigned short*)(Hbuf);              // [N, 2D]
    unsigned short* Yb  = (unsigned short*)(Hbuf + 2 * SLAB);
    unsigned short* Y2b = (unsigned short*)(Hbuf + 3 * SLAB);
    unsigned short* Hh16 = (unsigned short*)(Hbuf);             // [N, FFN]

    dim3 blk(256);
    auto T = [&](const float* s, unsigned short* d, int K, int N) {
        transpose_to_bf16<<<dim3((unsigned)((size_t)K * N / 256)), blk, 0,
                            stream>>>(s, d, K, N);
    };
    T(ff1_w1, wt_ff1_w1, Dm, FFNm);
    T(ff1_w2, wt_ff1_w2, FFNm, Dm);
    T(ff2_w1, wt_ff2_w1, Dm, FFNm);
    T(ff2_w2, wt_ff2_w2, FFNm, Dm);
    T(wq, wt_q, Dm, Dm);
    T(wk, wt_k, Dm, Dm);
    T(wv, wt_v, Dm, Dm);
    T(wo, wt_o, Dm, Dm);
    T(pw1_w, wt_pw1, Dm, 2 * Dm);
    T(pw2_w, wt_pw2, Dm, Dm);

    const dim3 lnGrid(Nt / 8);
    auto gemmGrid = [](int N) { return dim3(N / 128, Nt / 128); };

    // ---- FFN1 (half residual) ----
    layernorm_kernel<true><<<lnGrid, blk, 0, stream>>>(x, ff1_ln_g, ff1_ln_b,
                                                       xln, nullptr);
    gemm_bf16_kernel<EPI_GELU_BF16><<<gemmGrid(FFNm), blk, 0, stream>>>(
        xln, wt_ff1_w1, ff1_b1, nullptr, 0.f, nullptr, Hh16, FFNm, Dm);
    gemm_bf16_kernel<EPI_RESID_F32><<<gemmGrid(Dm), blk, 0, stream>>>(
        Hh16, wt_ff1_w2, ff1_b2, x, 0.5f, X1, nullptr, Dm, FFNm);

    // ---- Local attention ----
    layernorm_kernel<true><<<lnGrid, blk, 0, stream>>>(X1, attn_ln_g,
                                                       attn_ln_b, xln, nullptr);
    gemm_bf16_kernel<EPI_QK_HEAD><<<gemmGrid(Dm), blk, 0, stream>>>(
        xln, wt_q, bq, nullptr, 0.f, nullptr, Qb, Dm, Dm);
    gemm_bf16_kernel<EPI_QK_HEAD><<<gemmGrid(Dm), blk, 0, stream>>>(
        xln, wt_k, bk, nullptr, 0.f, nullptr, Kb, Dm, Dm);
    gemm_bf16_kernel<EPI_V_HEADT><<<gemmGrid(Dm), blk, 0, stream>>>(
        xln, wt_v, bvv, nullptr, 0.f, nullptr, VTb, Dm, Dm);
    local_attn_kernel<<<dim3(Ss / CTXs, Hh, Bb), blk, 0, stream>>>(
        Qb, Kb, VTb, rel_bias, AOb);
    gemm_bf16_kernel<EPI_RESID_F32><<<gemmGrid(Dm), blk, 0, stream>>>(
        AOb, wt_o, bo, X1, 1.f, X2, nullptr, Dm, Dm);

    // ---- Conv module ----
    layernorm_kernel<true><<<lnGrid, blk, 0, stream>>>(X2, conv_ln_g,
                                                       conv_ln_b, xln, nullptr);
    gemm_bf16_kernel<EPI_BF16><<<gemmGrid(2 * Dm), blk, 0, stream>>>(
        xln, wt_pw1, pw1_b, nullptr, 0.f, nullptr, G1, 2 * Dm, Dm);
    glu_kernel<<<dim3((unsigned)((size_t)Nt * Dm / 256)), blk, 0, stream>>>(
        G1, Yb);
    dwconv_bn_silu_kernel<<<dim3((unsigned)((size_t)Nt * Dm / 256)), blk, 0,
                            stream>>>(Yb, dw_w, dw_b, bn_g, bn_b, bn_m, bn_v,
                                      Y2b);
    gemm_bf16_kernel<EPI_RESID_F32><<<gemmGrid(Dm), blk, 0, stream>>>(
        Y2b, wt_pw2, pw2_b, X2, 1.f, X1, nullptr, Dm, Dm);

    // ---- FFN2 (half residual) ----
    layernorm_kernel<true><<<lnGrid, blk, 0, stream>>>(X1, ff2_ln_g, ff2_ln_b,
                                                       xln, nullptr);
    gemm_bf16_kernel<EPI_GELU_BF16><<<gemmGrid(FFNm), blk, 0, stream>>>(
        xln, wt_ff2_w1, ff2_b1, nullptr, 0.f, nullptr, Hh16, FFNm, Dm);
    gemm_bf16_kernel<EPI_RESID_F32><<<gemmGrid(Dm), blk, 0, stream>>>(
        Hh16, wt_ff2_w2, ff2_b2, X1, 0.5f, X2, nullptr, Dm, FFNm);

    // ---- Final LayerNorm -> d_out (f32) ----
    layernorm_kernel<false><<<lnGrid, blk, 0, stream>>>(
        X2, fin_ln_g, fin_ln_b, nullptr, (float*)d_out);
}
